// GIPAConv_25735444038232
// MI455X (gfx1250) — matrix operand order repaired and checked by reading, not compiled, not run on hardware
//
#include <hip/hip_runtime.h>

#define N_NODES   100000
#define N_EDGES   1600000
#define IN_FEATS  256
#define EDGE_FEATS 8
#define N_HEADS   4
#define OUT_FEATS 64
#define HF        256          // N_HEADS * OUT_FEATS
#define NEG_SLOPE 0.2f

typedef __bf16 bf16;
typedef __attribute__((ext_vector_type(16))) bf16  v16bf;
typedef __attribute__((ext_vector_type(8)))  float v8f;
typedef __attribute__((ext_vector_type(4)))  bf16  bf16x4;

// ---------------------------------------------------------------------------
// monotone float <-> uint encoding for atomicMax-based segment max
// ---------------------------------------------------------------------------
__device__ __forceinline__ unsigned fenc(float f) {
    unsigned u = __float_as_uint(f);
    return (u & 0x80000000u) ? ~u : (u | 0x80000000u);
}
__device__ __forceinline__ float fdec(unsigned k) {
    unsigned u = (k & 0x80000000u) ? (k & 0x7FFFFFFFu) : ~k;
    return __uint_as_float(u);
}
#define ENC_NEG_INF 0x007FFFFFu   // fenc(-inf)

// ---------------------------------------------------------------------------
// workspace init: zero h-accumulator + denom, seed segment-max with -inf
// ---------------------------------------------------------------------------
__global__ void k_init(float* __restrict__ hbuf, float* __restrict__ denom,
                       unsigned* __restrict__ emax) {
    int i = blockIdx.x * blockDim.x + threadIdx.x;
    if (i < N_NODES * HF) hbuf[i] = 0.0f;
    if (i < N_NODES * N_HEADS) { denom[i] = 0.0f; emax[i] = ENC_NEG_INF; }
}

// fp32 -> bf16 convert
__global__ void k_cvt_bf16(const float* __restrict__ x, bf16* __restrict__ y, int n) {
    int i = blockIdx.x * blockDim.x + threadIdx.x;
    if (i < n) y[i] = (bf16)x[i];
}

// W[HF][K] fp32 -> Wt[K][HF] bf16 (pre-transposed so GEMM B tiles are contiguous)
__global__ void k_transpose_cvt(const float* __restrict__ W, bf16* __restrict__ Wt) {
    int i = blockIdx.x * blockDim.x + threadIdx.x;
    if (i < HF * HF) {
        int n = i >> 8, k = i & 255;
        Wt[k * HF + n] = (bf16)W[n * HF + k];
    }
}

// ---------------------------------------------------------------------------
// bf16 WMMA GEMM:  C[M x 256] = A[M x 256] * Bt[256 x 256] (+bias) (+addend)
// block tile 128(M) x 256(N), K stepped by 32 through LDS; 8 waves, each
// owning a 16 x 256 slab (16 accumulators of v8f).
// ---------------------------------------------------------------------------
__global__ __launch_bounds__(256) void k_gemm_bf16(
        const bf16* __restrict__ A, const bf16* __restrict__ Bt,
        const float* __restrict__ bias, const float* __restrict__ addend,
        float* __restrict__ C, int M) {
    __shared__ bf16 As[128][40];    // 32 cols + pad
    __shared__ bf16 Bs[32][264];    // 256 cols + pad

    const int tid  = threadIdx.x;
    const int wave = tid >> 5;
    const int lane = tid & 31;
    const int m0   = blockIdx.x * 128;
    const int moff = wave * 16;

    const v8f vz = {0.f, 0.f, 0.f, 0.f, 0.f, 0.f, 0.f, 0.f};
    v8f acc[16];
#pragma unroll
    for (int nt = 0; nt < 16; ++nt) acc[nt] = vz;

    const bf16 bz = (bf16)0.0f;
    const bf16x4 vz4 = {bz, bz, bz, bz};

    for (int k0 = 0; k0 < 256; k0 += 32) {
        // ---- stage A tile (128x32) : 1024 4-elt chunks, 4 per thread
#pragma unroll
        for (int t = 0; t < 4; ++t) {
            int c  = tid + t * 256;
            int r  = c >> 3;
            int c4 = (c & 7) * 4;
            int rg = m0 + r;
            bf16x4 v = vz4;
            if (rg < M) v = *(const bf16x4*)(A + (size_t)rg * 256 + k0 + c4);
            *(bf16x4*)&As[r][c4] = v;
        }
        // ---- stage B tile (32x256) : 2048 4-elt chunks, 8 per thread
#pragma unroll
        for (int t = 0; t < 8; ++t) {
            int c  = tid + t * 256;
            int r  = c >> 6;
            int c4 = (c & 63) * 4;
            *(bf16x4*)&Bs[r][c4] = *(const bf16x4*)(Bt + (size_t)(k0 + r) * 256 + c4);
        }
        // prefetch next K tile into cache while we compute (global_prefetch_b8)
        if (k0 + 32 < 256) {
            __builtin_prefetch(Bt + (size_t)(k0 + 32 + (tid >> 3)) * 256 + (tid & 7) * 32, 0, 1);
        }
        __syncthreads();

        // ---- A fragment (16x32 bf16, ISA wave32 layout)
        const int m  = lane & 15;
        const int hh = lane >> 4;
        v16bf afrag;
#pragma unroll
        for (int i = 0; i < 8; ++i) {
            int kk = ((i < 4) ? 0 : 16) + hh * 8 + (i & 3) * 2;
            afrag[2 * i]     = As[moff + m][kk];
            afrag[2 * i + 1] = As[moff + m][kk + 1];
        }
        // ---- 16 WMMAs across the 256-wide N slab (B 32x16: lane = N column, elements = K)
#pragma unroll
        for (int nt = 0; nt < 16; ++nt) {
            v16bf bfrag;
#pragma unroll
            for (int j = 0; j < 8; ++j) {
                int kb = ((j < 4) ? 0 : 16) + hh * 8 + (j & 3) * 2;
                bfrag[2 * j]     = Bs[kb][nt * 16 + m];
                bfrag[2 * j + 1] = Bs[kb + 1][nt * 16 + m];
            }
            acc[nt] = __builtin_amdgcn_wmma_f32_16x16x32_bf16(
                false, afrag, false, bfrag, (short)0, acc[nt], false, false);
        }
        __syncthreads();
    }

    // ---- epilogue: C layout m = 8*(lane>=16)+r , n = lane&15
    const int n  = lane & 15;
    const int mb = (lane >> 4) * 8;
#pragma unroll
    for (int nt = 0; nt < 16; ++nt) {
        int col = nt * 16 + n;
        float bv = bias ? bias[col] : 0.0f;
#pragma unroll
        for (int r = 0; r < 8; ++r) {
            int rg = m0 + moff + mb + r;
            if (rg < M) {
                size_t o = (size_t)rg * 256 + col;
                float v = acc[nt][r] + bv;
                if (addend) v += addend[o];
                C[o] = v;
            }
        }
    }
}

// ---------------------------------------------------------------------------
// attn_src / attn_dst : one wave per node, 8 dot products of length 256
// ---------------------------------------------------------------------------
__global__ __launch_bounds__(256) void k_node_attn(
        const float* __restrict__ X,
        const float* __restrict__ Was, const float* __restrict__ Wad,
        float* __restrict__ asrc, float* __restrict__ adst) {
    __shared__ float ws[N_HEADS][256];
    __shared__ float wd[N_HEADS][256];
    const int tid = threadIdx.x;
    for (int i = tid; i < N_HEADS * 256; i += 256) {
        ws[i >> 8][i & 255] = Was[i];
        wd[i >> 8][i & 255] = Wad[i];
    }
    __syncthreads();

    const int wave = tid >> 5, lane = tid & 31;
    const int node = blockIdx.x * 8 + wave;
    if (node >= N_NODES) return;

    float acc[8] = {0.f, 0.f, 0.f, 0.f, 0.f, 0.f, 0.f, 0.f};
    const float* xr = X + (size_t)node * 256 + lane * 8;
#pragma unroll
    for (int j = 0; j < 8; ++j) {
        float x = xr[j];
        int k = lane * 8 + j;
#pragma unroll
        for (int h = 0; h < N_HEADS; ++h) {
            acc[h]     += x * ws[h][k];
            acc[4 + h] += x * wd[h][k];
        }
    }
#pragma unroll
    for (int off = 16; off >= 1; off >>= 1)
#pragma unroll
        for (int t = 0; t < 8; ++t) acc[t] += __shfl_xor(acc[t], off, 32);

    if (lane < N_HEADS) {
        asrc[node * N_HEADS + lane] = acc[lane];
        adst[node * N_HEADS + lane] = acc[4 + lane];
    }
}

// ---------------------------------------------------------------------------
// per-edge logits + leaky_relu + segment-max via encoded atomicMax
// ---------------------------------------------------------------------------
__global__ void k_edge_logits(
        const float* __restrict__ fe, const int* __restrict__ src,
        const int* __restrict__ dst, const float* __restrict__ Wae,
        const float* __restrict__ asrc, const float* __restrict__ adst,
        float* __restrict__ elog, unsigned* __restrict__ emax) {
    int e = blockIdx.x * blockDim.x + threadIdx.x;
    if (e >= N_EDGES) return;
    const int s = src[e], d = dst[e];
    float f[EDGE_FEATS];
#pragma unroll
    for (int j = 0; j < EDGE_FEATS; ++j) f[j] = fe[(size_t)e * EDGE_FEATS + j];
#pragma unroll
    for (int h = 0; h < N_HEADS; ++h) {
        float v = asrc[s * N_HEADS + h] + adst[d * N_HEADS + h];
#pragma unroll
        for (int j = 0; j < EDGE_FEATS; ++j) v += f[j] * Wae[h * EDGE_FEATS + j];
        v = (v > 0.0f) ? v : NEG_SLOPE * v;
        elog[(size_t)e * N_HEADS + h] = v;
        atomicMax(&emax[d * N_HEADS + h], fenc(v));
    }
}

// exp(e - max) in place + denominator segment-sum
__global__ void k_edge_exp(const int* __restrict__ dst,
                           const unsigned* __restrict__ emax,
                           float* __restrict__ elog,
                           float* __restrict__ denom) {
    int i = blockIdx.x * blockDim.x + threadIdx.x;
    if (i >= N_EDGES * N_HEADS) return;
    int e = i >> 2, h = i & 3;
    int d = dst[e];
    float v = expf(elog[i] - fdec(emax[d * N_HEADS + h]));
    elog[i] = v;
    atomicAdd(&denom[d * N_HEADS + h], v);
}

// ---------------------------------------------------------------------------
// message + scatter: one wave per edge; 256 floats gathered + atomically added
// (feat_src_fc and hbuf are each ~102MB: both L2-resident on 192MB L2)
// ---------------------------------------------------------------------------
__global__ __launch_bounds__(256) void k_message(
        const int* __restrict__ src, const int* __restrict__ dst,
        const float* __restrict__ eexp, const float* __restrict__ denom,
        const float* __restrict__ fsrc_fc, float* __restrict__ hbuf) {
    const int wave = threadIdx.x >> 5, lane = threadIdx.x & 31;
    const int e = blockIdx.x * 8 + wave;
    if (e >= N_EDGES) return;
    const int s = src[e], d = dst[e];
    float coef[N_HEADS];
#pragma unroll
    for (int h = 0; h < N_HEADS; ++h)
        coef[h] = eexp[(size_t)e * N_HEADS + h] / denom[d * N_HEADS + h];
    const float* frow = fsrc_fc + (size_t)s * HF;
    float* hrow = hbuf + (size_t)d * HF;
#pragma unroll
    for (int t = 0; t < 8; ++t) {
        int idx = t * 32 + lane;            // head = t>>1 (64 feats per head)
        atomicAdd(&hrow[idx], frow[idx] * coef[t >> 1]);
    }
}

// ---------------------------------------------------------------------------
// per-(node,head) batchnorm-like normalize over 64 feats; one wave per row,
// writes bf16 directly as the A operand of the final WMMA GEMM
// ---------------------------------------------------------------------------
__global__ __launch_bounds__(256) void k_norm(
        const float* __restrict__ hbuf, const float* __restrict__ scale,
        const float* __restrict__ offs, bf16* __restrict__ Hbf) {
    const int wave = threadIdx.x >> 5, lane = threadIdx.x & 31;
    const int row = blockIdx.x * 8 + wave;     // over N_NODES * N_HEADS
    if (row >= N_NODES * N_HEADS) return;
    const int h = row & 3;
    const size_t base = (size_t)row * OUT_FEATS;
    float x0 = hbuf[base + lane];
    float x1 = hbuf[base + 32 + lane];
    float s  = x0 + x1;
    float ss = x0 * x0 + x1 * x1;
#pragma unroll
    for (int off = 16; off >= 1; off >>= 1) {
        s  += __shfl_xor(s,  off, 32);
        ss += __shfl_xor(ss, off, 32);
    }
    float mean = s * (1.0f / 64.0f);
    float var  = ss * (1.0f / 64.0f) - mean * mean + 1e-9f;
    float rinv = rsqrtf(var);
    float y0 = (x0 - mean) * scale[h * 64 + lane]      * rinv + offs[h * 64 + lane];
    float y1 = (x1 - mean) * scale[h * 64 + 32 + lane] * rinv + offs[h * 64 + 32 + lane];
    Hbf[base + lane]      = (bf16)y0;
    Hbf[base + 32 + lane] = (bf16)y1;
}

// ---------------------------------------------------------------------------
extern "C" void kernel_launch(void* const* d_in, const int* in_sizes, int n_in,
                              void* d_out, int out_size, void* d_ws, size_t ws_size,
                              hipStream_t stream) {
    const float* feat_src    = (const float*)d_in[0];
    const float* feat_edge   = (const float*)d_in[1];
    const int*   src         = (const int*)d_in[2];
    const int*   dst         = (const int*)d_in[3];
    const float* W_src       = (const float*)d_in[4];
    const float* W_attn_src  = (const float*)d_in[5];
    const float* W_attn_dst  = (const float*)d_in[6];
    const float* W_attn_edge = (const float*)d_in[7];
    const float* W_agg       = (const float*)d_in[8];
    const float* b_agg       = (const float*)d_in[9];
    const float* W_dst       = (const float*)d_in[10];
    const float* b_dst       = (const float*)d_in[11];
    const float* scale       = (const float*)d_in[12];
    const float* offset      = (const float*)d_in[13];
    float* out = (float*)d_out;

    // carve workspace (256B aligned)
    char* ws = (char*)d_ws;
    auto carve = [&](size_t bytes) -> char* {
        char* p = ws;
        ws += (bytes + 255) & ~(size_t)255;
        return p;
    };
    bf16*     Xbf     = (bf16*)carve((size_t)N_NODES * HF * sizeof(bf16));
    bf16*     Hbf     = (bf16*)carve((size_t)N_NODES * HF * sizeof(bf16));
    bf16*     WsT     = (bf16*)carve((size_t)HF * HF * sizeof(bf16));
    bf16*     WdT     = (bf16*)carve((size_t)HF * HF * sizeof(bf16));
    bf16*     WaT     = (bf16*)carve((size_t)HF * HF * sizeof(bf16));
    float*    fsrc_fc = (float*)carve((size_t)N_NODES * HF * sizeof(float));
    float*    fdst_fc = (float*)carve((size_t)N_NODES * HF * sizeof(float));
    float*    hbuf    = (float*)carve((size_t)N_NODES * HF * sizeof(float));
    float*    asrc    = (float*)carve((size_t)N_NODES * N_HEADS * sizeof(float));
    float*    adst    = (float*)carve((size_t)N_NODES * N_HEADS * sizeof(float));
    float*    denom   = (float*)carve((size_t)N_NODES * N_HEADS * sizeof(float));
    unsigned* emax    = (unsigned*)carve((size_t)N_NODES * N_HEADS * sizeof(unsigned));
    float*    elog    = (float*)carve((size_t)N_EDGES * N_HEADS * sizeof(float));

    const int B = 256;
    // init accumulators / seg-max
    k_init<<<(N_NODES * HF + B - 1) / B, B, 0, stream>>>(hbuf, denom, emax);
    // precision staging
    k_cvt_bf16<<<(N_NODES * HF + B - 1) / B, B, 0, stream>>>(feat_src, Xbf, N_NODES * HF);
    k_transpose_cvt<<<(HF * HF + B - 1) / B, B, 0, stream>>>(W_src, WsT);
    k_transpose_cvt<<<(HF * HF + B - 1) / B, B, 0, stream>>>(W_dst, WdT);
    k_transpose_cvt<<<(HF * HF + B - 1) / B, B, 0, stream>>>(W_agg, WaT);

    const int gemm_grid = (N_NODES + 127) / 128;
    // feat_src_fc = X @ W_src.T
    k_gemm_bf16<<<gemm_grid, B, 0, stream>>>(Xbf, WsT, nullptr, nullptr, fsrc_fc, N_NODES);
    // feat_dst_fc = X @ W_dst.T + b_dst
    k_gemm_bf16<<<gemm_grid, B, 0, stream>>>(Xbf, WdT, b_dst, nullptr, fdst_fc, N_NODES);

    // attention scalars
    k_node_attn<<<(N_NODES + 7) / 8, B, 0, stream>>>(feat_src, W_attn_src, W_attn_dst, asrc, adst);
    k_edge_logits<<<(N_EDGES + B - 1) / B, B, 0, stream>>>(feat_edge, src, dst, W_attn_edge,
                                                           asrc, adst, elog, emax);
    k_edge_exp<<<(N_EDGES * N_HEADS + B - 1) / B, B, 0, stream>>>(dst, emax, elog, denom);

    // gather * a -> scatter-add
    k_message<<<(N_EDGES + 7) / 8, B, 0, stream>>>(src, dst, elog, denom, fsrc_fc, hbuf);

    // normalize + bf16
    k_norm<<<(N_NODES * N_HEADS + 7) / 8, B, 0, stream>>>(hbuf, scale, offset, Hbf);

    // out = Hnorm @ W_agg.T + b_agg + feat_dst_fc   (fused epilogue, writes d_out)
    k_gemm_bf16<<<gemm_grid, B, 0, stream>>>(Hbf, WaT, b_agg, fdst_fc, out, N_NODES);
}
